// MotionDetectionLSTM_31516470018558
// MI455X (gfx1250) — compile-verified, matrix operand order checked by reading, not run on hardware
//
#include <hip/hip_runtime.h>
#include <hip/hip_bf16.h>
#include <math.h>

typedef __attribute__((ext_vector_type(16))) _Float16 v16h;
typedef __attribute__((ext_vector_type(8)))  _Float16 v8h;
typedef __attribute__((ext_vector_type(8)))  float    v8f;

#define BT   512      // B*T = 16*32
#define TDIM 32
#define BDIM 16
#define HID  100
#define FEAT 2048
#define NG   400      // 4*HID

// ---- workspace layout (bytes) ----
#define P1_OFF 0                              // p1: 512*8*32*32 f32 = 16 MiB
#define FE_OFF (512u * 8192u * 4u)            // feats f16: 512*2048
#define WI_OFF (FE_OFF + 512u * 2048u * 2u)   // w_ih f16: 400*2048
#define G_OFF  (WI_OFF + 400u * 2048u * 2u)   // gates_in f32: 512*400

__device__ __forceinline__ float sigm(float x) { return 1.0f / (1.0f + __expf(-x)); }

// =====================================================================
// Kernel 1: avgpool2 -> conv1(5x5, 1->8, pad 2) -> ReLU -> avgpool4
// one block per frame; pooled 128x128 input staged in dynamic LDS (64KB)
// =====================================================================
__global__ void k_stage1(const float* __restrict__ x,
                         const float* __restrict__ w1,
                         const float* __restrict__ b1,
                         float* __restrict__ p1) {
    extern __shared__ float s_x0[];               // 128*128 floats = 64 KB
    const int bt  = blockIdx.x;
    const int tid = threadIdx.x;                  // 256 threads
    const float* img = x + (size_t)bt * 256 * 256;

    // avgpool2: 256x256 -> 128x128 into LDS
    for (int i = 0; i < 64; ++i) {
        int p  = tid + i * 256;
        int py = p >> 7, px = p & 127;
        const float* r0 = img + (2 * py) * 256 + 2 * px;
        s_x0[p] = 0.25f * (r0[0] + r0[1] + r0[256] + r0[257]);
    }
    __syncthreads();

    float* outimg = p1 + (size_t)bt * 8 * 32 * 32;
    for (int c = 0; c < 8; ++c) {
        float wr[25];
#pragma unroll
        for (int k = 0; k < 25; ++k) wr[k] = w1[c * 25 + k];
        const float bias = b1[c];
#pragma unroll
        for (int i = 0; i < 4; ++i) {             // 1024 pooled outputs / 256 thr
            int o  = tid + i * 256;
            int oy = o >> 5, ox = o & 31;
            float sum = 0.0f;
            for (int sy = 0; sy < 4; ++sy) {
                int y = 4 * oy + sy;
                for (int sx = 0; sx < 4; ++sx) {
                    int x0 = 4 * ox + sx;
                    float acc = bias;
#pragma unroll
                    for (int dy = 0; dy < 5; ++dy) {
                        int iy = y + dy - 2;
                        if ((unsigned)iy >= 128u) continue;
#pragma unroll
                        for (int dx = 0; dx < 5; ++dx) {
                            int ix = x0 + dx - 2;
                            if ((unsigned)ix >= 128u) continue;
                            acc = fmaf(wr[dy * 5 + dx], s_x0[iy * 128 + ix], acc);
                        }
                    }
                    sum += fmaxf(acc, 0.0f);
                }
            }
            outimg[c * 1024 + o] = sum * (1.0f / 16.0f);
        }
    }
}

// =====================================================================
// Kernel 2: conv2(5x5, 8->32, pad 2) -> ReLU -> avgpool4 -> feats (f16)
// one block per frame; p1 (32KB) + conv2 weights (25.6KB) in dynamic LDS
// =====================================================================
__global__ void k_stage2(const float* __restrict__ p1,
                         const float* __restrict__ w2,
                         const float* __restrict__ b2,
                         _Float16* __restrict__ feats) {
    extern __shared__ float smem[];
    float* s_p1 = smem;                 // 8*32*32 = 8192 floats
    float* s_w2 = smem + 8192;          // 32*8*25 = 6400 floats
    float* s_b2 = s_w2 + 6400;          // 32 floats
    const int bt  = blockIdx.x;
    const int tid = threadIdx.x;        // 256 threads

    const float* src = p1 + (size_t)bt * 8192;
    for (int i = 0; i < 32; ++i) s_p1[tid + i * 256] = src[tid + i * 256];
    for (int i = 0; i < 25; ++i) s_w2[tid + i * 256] = w2[tid + i * 256];
    if (tid < 32) s_b2[tid] = b2[tid];
    __syncthreads();

    _Float16* frow = feats + (size_t)bt * FEAT;
#pragma unroll
    for (int i = 0; i < 8; ++i) {       // 2048 outputs / 256 thr
        int o   = tid + i * 256;
        int oc  = o >> 6;
        int rem = o & 63;
        int oy  = rem >> 3, ox = rem & 7;
        const float* wc = s_w2 + oc * 200;
        const float bias = s_b2[oc];
        float sum = 0.0f;
        for (int sy = 0; sy < 4; ++sy) {
            int y = 4 * oy + sy;
            for (int sx = 0; sx < 4; ++sx) {
                int x0 = 4 * ox + sx;
                float acc = bias;
                for (int ic = 0; ic < 8; ++ic) {
                    const float* pc = s_p1 + ic * 1024;
                    const float* wi = wc + ic * 25;
#pragma unroll
                    for (int dy = 0; dy < 5; ++dy) {
                        int iy = y + dy - 2;
                        if ((unsigned)iy >= 32u) continue;
#pragma unroll
                        for (int dx = 0; dx < 5; ++dx) {
                            int ix = x0 + dx - 2;
                            if ((unsigned)ix >= 32u) continue;
                            acc = fmaf(wi[dy * 5 + dx], pc[iy * 32 + ix], acc);
                        }
                    }
                }
                sum += fmaxf(acc, 0.0f);
            }
        }
        frow[o] = (_Float16)(sum * (1.0f / 16.0f));
    }
}

// =====================================================================
// Kernel 3: convert w_ih f32 -> f16 (same [400][2048] layout)
// =====================================================================
__global__ void k_cvt(const float* __restrict__ w, _Float16* __restrict__ wh, int n) {
    int i = blockIdx.x * 256 + threadIdx.x;
    if (i < n) wh[i] = (_Float16)w[i];
}

// =====================================================================
// Kernel 4: WMMA GEMM  G[512,400] = feats[512,2048] x w_ih^T + b_ih
// one wave per 16x16 tile; K=2048 in 64 steps of v_wmma_f32_16x16x32_f16
// A: 16x32 f16 (M=lane&15, K chunks by lane>>4)  B: 32x16 f16 (N=lane&15)
// =====================================================================
__global__ void k_gemm(const _Float16* __restrict__ feats,
                       const _Float16* __restrict__ wih,
                       const float* __restrict__ b_ih,
                       float* __restrict__ G) {
    const int wave = threadIdx.x >> 5;                 // 8 waves / block
    const int lane = threadIdx.x & 31;
    const int tile = blockIdx.x * 8 + wave;            // 800 tiles = 32 x 25
    const int tm   = tile / 25;                        // M tile (0..31)
    const int tn   = tile % 25;                        // N tile (0..24)
    const int mn   = lane & 15;
    const int h    = lane >> 4;

    const _Float16* arow = feats + (size_t)(tm * 16 + mn) * FEAT;  // A row (M)
    const _Float16* brow = wih   + (size_t)(tn * 16 + mn) * FEAT;  // B col (N)

    v8f acc = {};
    for (int k0 = 0; k0 < FEAT; k0 += 32) {
        __builtin_prefetch((const void*)(arow + k0 + 256), 0, 1);  // global_prefetch_b8
        const v8h a0 = *(const v8h*)(arow + k0 + 8 * h);
        const v8h a1 = *(const v8h*)(arow + k0 + 16 + 8 * h);
        const v8h b0 = *(const v8h*)(brow + k0 + 8 * h);
        const v8h b1 = *(const v8h*)(brow + k0 + 16 + 8 * h);
        v16h A, Bm;
#pragma unroll
        for (int e = 0; e < 8; ++e) { A[e] = a0[e]; A[8 + e] = a1[e]; Bm[e] = b0[e]; Bm[8 + e] = b1[e]; }
        acc = __builtin_amdgcn_wmma_f32_16x16x32_f16(
            /*neg_a=*/false, A, /*neg_b=*/false, Bm,
            /*c_mod=*/(short)0, acc, /*reuse_a=*/false, /*reuse_b=*/false);
    }

    const int n = tn * 16 + mn;
    const float bias = b_ih[n];
#pragma unroll
    for (int r = 0; r < 8; ++r) {                      // C/D: M = r + 8*h, N = lane&15
        int m = tm * 16 + r + 8 * h;
        G[(size_t)m * NG + n] = acc[r] + bias;
    }
}

// =====================================================================
// Kernel 5: recurrent LSTM over batch per timestep + final linear
// one block per t (32 blocks, 128 thr); h/c/gates in LDS
// =====================================================================
__global__ void k_lstm(const float* __restrict__ G,
                       const float* __restrict__ w_hh,
                       const float* __restrict__ b_hh,
                       const float* __restrict__ lin_w,
                       float* __restrict__ out) {
    __shared__ float hS[HID], cS[HID], gS[NG];
    const int t   = blockIdx.x;
    const int tid = threadIdx.x;          // 128 threads

    if (tid < HID) { hS[tid] = 0.0f; cS[tid] = 0.0f; }
    __syncthreads();

    for (int b = 0; b < BDIM; ++b) {
        const int r = b * TDIM + t;       // feats row order [B,T]
        for (int j = tid; j < NG; j += 128) {
            float acc = G[(size_t)r * NG + j] + b_hh[j];
            const float* wr = w_hh + (size_t)j * HID;
            for (int k = 0; k < HID; ++k) acc = fmaf(hS[k], wr[k], acc);
            gS[j] = acc;
        }
        __syncthreads();
        if (tid < HID) {
            float ig = gS[tid], fg = gS[HID + tid], gg = gS[2 * HID + tid], og = gS[3 * HID + tid];
            float cc = sigm(fg) * cS[tid] + sigm(ig) * tanhf(gg);
            float hh = sigm(og) * tanhf(cc);
            cS[tid] = cc; hS[tid] = hh;
        }
        __syncthreads();
        if (tid == 0) {
            float s = 0.0f;
            for (int k = 0; k < HID; ++k) s = fmaf(hS[k], lin_w[k], s);
            out[b * TDIM + t] = s;        // output [B,T,1]
        }
        __syncthreads();
    }
}

extern "C" void kernel_launch(void* const* d_in, const int* in_sizes, int n_in,
                              void* d_out, int out_size, void* d_ws, size_t ws_size,
                              hipStream_t stream) {
    const float* x      = (const float*)d_in[0];
    const float* c1w    = (const float*)d_in[1];
    const float* c1b    = (const float*)d_in[2];
    const float* c2w    = (const float*)d_in[3];
    const float* c2b    = (const float*)d_in[4];
    const float* w_ih   = (const float*)d_in[5];
    const float* w_hh   = (const float*)d_in[6];
    const float* b_ih   = (const float*)d_in[7];
    const float* b_hh   = (const float*)d_in[8];
    const float* lin_w  = (const float*)d_in[9];
    float* out = (float*)d_out;

    char* ws = (char*)d_ws;
    float*    p1    = (float*)(ws + P1_OFF);
    _Float16* feats = (_Float16*)(ws + FE_OFF);
    _Float16* wihh  = (_Float16*)(ws + WI_OFF);
    float*    G     = (float*)(ws + G_OFF);

    // stage 1: pool2 + conv1 + relu + pool4 (64 KB dynamic LDS)
    k_stage1<<<BT, 256, 128 * 128 * 4, stream>>>(x, c1w, c1b, p1);
    // stage 2: conv2 + relu + pool4 -> f16 feats (~57 KB dynamic LDS)
    k_stage2<<<BT, 256, (8192 + 6400 + 32) * 4, stream>>>(p1, c2w, c2b, feats);
    // w_ih -> f16
    k_cvt<<<(NG * FEAT + 255) / 256, 256, 0, stream>>>(w_ih, wihh, NG * FEAT);
    // WMMA input-gate GEMM: 800 tiles, 8 waves/block
    k_gemm<<<100, 256, 0, stream>>>(feats, wihh, b_ih, G);
    // recurrent scan + linear head
    k_lstm<<<TDIM, 128, 0, stream>>>(G, w_hh, b_hh, lin_w, out);
}
